// YOLOLoss_32341103739070
// MI455X (gfx1250) — compile-verified, matrix operand order checked by reading, not run on hardware
//
#include <hip/hip_runtime.h>
#include <hip/hip_bf16.h>

// ---------------------------------------------------------------------------
// YOLO loss (ind=2 branch): bs=32, A=3, 52x52 grid, 80 classes.
//  K1: per-target anchor matching -> obj map (atomicMax => deterministic)
//  K2: per-cell pass: conf BCE + ignore-IoU vs 50 LDS-staged targets
//      (async global->LDS copy on gfx1250) + GIoU/cls BCE at obj cells;
//      fixed-order block reduction -> 5 partial sums per block (SoA)
//  K3: single-wave WMMA (f32_16x16x32_f16, B=ones) tensor-core reduction of
//      the 1056x5 partials, then final loss formula. Fully deterministic.
// ---------------------------------------------------------------------------

typedef __attribute__((ext_vector_type(16))) _Float16 v16h;
typedef __attribute__((ext_vector_type(8)))  float    v8f;
typedef __attribute__((ext_vector_type(4)))  float    v4f_;

#define BSZ   32
#define NA    3           // anchors per head (mask [0,1,2] for ind=2)
#define GH    52
#define GW    52
#define GHW   (GH*GW)     // 2704
#define NT    50
#define NC    80
#define NATTR 85
#define TILE  256
#define TILES_PER_BA ((GHW + TILE - 1)/TILE)   // 11
#define NBLK  (BSZ*NA*TILES_PER_BA)            // 1056 (== 11 * 96)
#define EPSF  1e-7f

#if defined(__HIP_DEVICE_COMPILE__) && \
    __has_builtin(__builtin_amdgcn_global_load_async_to_lds_b128) && \
    __has_builtin(__builtin_amdgcn_s_wait_asynccnt)
#define HAVE_ASYNC_LDS 1
// Builtin signature (from hipcc diagnostic): arg0 is AS1 pointer to
// 'int __attribute__((vector_size(16)))'; use matching typedefs.
typedef int v4i_ __attribute__((vector_size(4 * sizeof(int))));
typedef __attribute__((address_space(1))) v4i_* as1_v4i;
typedef __attribute__((address_space(3))) v4i_* as3_v4i;
#else
#define HAVE_ASYNC_LDS 0
#endif

__device__ __forceinline__ float sigmoidf_(float x) {
  return 1.0f / (1.0f + __expf(-x));
}
__device__ __forceinline__ float bce_clip(float p, float t) {
  p = fminf(fmaxf(p, EPSF), 1.0f - EPSF);
  return -t * __logf(p) - (1.0f - t) * __logf(1.0f - p);
}
__device__ __forceinline__ float iou_xywh(float ax, float ay, float aw, float ah,
                                          float bx, float by, float bw, float bh) {
  float ax0 = ax - aw*0.5f, ax1 = ax + aw*0.5f;
  float ay0 = ay - ah*0.5f, ay1 = ay + ah*0.5f;
  float bx0 = bx - bw*0.5f, bx1 = bx + bw*0.5f;
  float by0 = by - bh*0.5f, by1 = by + bh*0.5f;
  float iw = fmaxf(fminf(ax1, bx1) - fmaxf(ax0, bx0), 0.0f);
  float ih = fmaxf(fminf(ay1, by1) - fmaxf(ay0, by0), 0.0f);
  float inter = iw * ih;
  float areaA = (ax1-ax0)*(ay1-ay0);
  float areaB = (bx1-bx0)*(by1-by0);
  return inter / (areaA + areaB - inter);
}
__device__ __forceinline__ float giou_xywh(float ax, float ay, float aw, float ah,
                                           float bx, float by, float bw, float bh) {
  float ax0 = ax - aw*0.5f, ax1 = ax + aw*0.5f;
  float ay0 = ay - ah*0.5f, ay1 = ay + ah*0.5f;
  float bx0 = bx - bw*0.5f, bx1 = bx + bw*0.5f;
  float by0 = by - bh*0.5f, by1 = by + bh*0.5f;
  float iw = fmaxf(fminf(ax1, bx1) - fmaxf(ax0, bx0), 0.0f);
  float ih = fmaxf(fminf(ay1, by1) - fmaxf(ay0, by0), 0.0f);
  float inter = iw * ih;
  float uni = aw*ah + bw*bh - inter;
  float iou = inter / uni;
  float ew = fmaxf(fmaxf(ax1, bx1) - fminf(ax0, bx0), 0.0f);
  float eh = fmaxf(fmaxf(ay1, by1) - fminf(ay0, by0), 0.0f);
  float enc = ew * eh;
  return iou - (enc - uni) / enc;
}

// --------------------- K1: target -> anchor matching -----------------------
__global__ void yolo_match_kernel(const float* __restrict__ tboxes,
                                  int* __restrict__ objmap) {
  int t = blockIdx.x * blockDim.x + threadIdx.x;
  if (t >= BSZ * NT) return;
  int b  = t / NT;
  int tt = t - b * NT;
  float bx = tboxes[t*4 + 0] * (float)GW;
  float by = tboxes[t*4 + 1] * (float)GH;
  float bw = tboxes[t*4 + 2] * (float)GW;
  float bh = tboxes[t*4 + 3] * (float)GH;
  // scaled anchors (stride 8)
  const float SA[9][2] = {
    {1.25f,1.625f},{2.0f,3.75f},{4.125f,2.875f},
    {3.75f,7.625f},{7.75f,5.625f},{7.375f,14.875f},
    {14.5f,11.25f},{19.5f,24.75f},{46.625f,40.75f}};
  float best = -1.0f; int bestn = 0;
  #pragma unroll
  for (int an = 0; an < 9; ++an) {
    float aw = SA[an][0], ah = SA[an][1];
    float inter = fminf(bw, aw) * fminf(bh, ah);
    float uni   = bw*bh + aw*ah - inter;
    float iou   = inter / uni;
    if (iou > best) { best = iou; bestn = an; }  // first-max tie-break
  }
  if (bestn < NA) {                               // mask [0,1,2] for ind=2
    int gi = (int)floorf(bx); gi = min(max(gi, 0), GW-1);
    int gj = (int)floorf(by); gj = min(max(gj, 0), GH-1);
    int cell = ((b*NA + bestn)*GH + gj)*GW + gi;
    atomicMax(&objmap[cell], tt + 1);             // deterministic dedup
  }
}

// --------------------- K2: per-cell loss pieces ----------------------------
__global__ void yolo_cell_kernel(const float* __restrict__ input,
                                 const float* __restrict__ tboxes,
                                 const int*   __restrict__ tcls,
                                 const int*   __restrict__ objmap,
                                 float* __restrict__ part) {
  __shared__ float s_t[NT*4];
  __shared__ float red[TILE];
  const int blk  = blockIdx.x;
  const int ba   = blk / TILES_PER_BA;
  const int tile = blk - ba * TILES_PER_BA;
  const int b    = ba / NA;
  const int a    = ba - b * NA;

  // Stage this batch's 50 target boxes into LDS (async copy on gfx1250).
#if HAVE_ASYNC_LDS
  if (threadIdx.x < NT) {
    const float* g = tboxes + (size_t)b*NT*4 + (size_t)threadIdx.x*4;
    __builtin_amdgcn_global_load_async_to_lds_b128(
        (as1_v4i)(void*)g, (as3_v4i)(void*)(&s_t[threadIdx.x*4]), 0, 0);
  }
  __builtin_amdgcn_s_wait_asynccnt(0);
  __syncthreads();
#else
  if (threadIdx.x < NT*4)
    s_t[threadIdx.x] = tboxes[(size_t)b*NT*4 + threadIdx.x];
  __syncthreads();
#endif

  float conf_l = 0.0f, conf_m = 0.0f, loc_s = 0.0f, cls_s = 0.0f, nobj = 0.0f;
  const int cell = tile * TILE + (int)threadIdx.x;
  if (cell < GHW) {
    const int j = cell / GW;
    const int i = cell - j * GW;
    const size_t base = ((size_t)(b*255 + a*NATTR)) * GHW + (size_t)cell;
    const float tx = input[base + (size_t)0*GHW];
    const float ty = input[base + (size_t)1*GHW];
    const float tw = input[base + (size_t)2*GHW];
    const float th = input[base + (size_t)3*GHW];
    const float tc = input[base + (size_t)4*GHW];
    const float aw = (a == 0) ? 1.25f  : ((a == 1) ? 2.0f  : 4.125f);
    const float ah = (a == 0) ? 1.625f : ((a == 1) ? 3.75f : 2.875f);
    const float px = sigmoidf_(tx) + (float)i;
    const float py = sigmoidf_(ty) + (float)j;
    const float pw = __expf(tw) * aw;
    const float ph = __expf(th) * ah;

    // ignore-threshold: max IoU of this pred box vs all 50 targets
    float maxiou = -1e30f;
    #pragma unroll 5
    for (int t = 0; t < NT; ++t) {
      float bx = s_t[t*4+0] * (float)GW;
      float by = s_t[t*4+1] * (float)GH;
      float bw = s_t[t*4+2] * (float)GW;
      float bh = s_t[t*4+3] * (float)GH;
      maxiou = fmaxf(maxiou, iou_xywh(px, py, pw, ph, bx, by, bw, bh));
    }

    const int omap = objmap[(size_t)(b*NA + a)*GHW + cell];
    const bool obj = (omap != 0);
    const float objf = obj ? 1.0f : 0.0f;
    const float cmask = obj ? 1.0f : ((maxiou > 0.5f) ? 0.0f : 1.0f);
    conf_l = bce_clip(sigmoidf_(tc), objf) * cmask;
    conf_m = cmask;
    nobj   = objf;

    if (obj) {
      const int t = omap - 1;
      const float bx = s_t[t*4+0] * (float)GW;
      const float by = s_t[t*4+1] * (float)GH;
      const float bw = s_t[t*4+2] * (float)GW;
      const float bh = s_t[t*4+3] * (float)GH;
      loc_s = 1.0f - giou_xywh(px, py, pw, ph, bx, by, bw, bh);
      const int cls = tcls[b*NT + t];
      float cs = 0.0f;
      for (int c2 = 0; c2 < NC; ++c2) {
        // reference applies BCE-with-clip to RAW class logits
        float p = input[base + (size_t)(5 + c2)*GHW];
        cs += bce_clip(p, (c2 == cls) ? 1.0f : 0.0f);
      }
      cls_s = cs;
    }
  }

  // fixed-order block reduction of the 5 accumulators
  float vals[5] = { conf_l, conf_m, loc_s, cls_s, nobj };
  for (int q = 0; q < 5; ++q) {
    red[threadIdx.x] = vals[q];
    __syncthreads();
    for (int s = TILE >> 1; s > 0; s >>= 1) {
      if (threadIdx.x < s) red[threadIdx.x] += red[threadIdx.x + s];
      __syncthreads();
    }
    if (threadIdx.x == 0) part[q*NBLK + blk] = red[0];
    __syncthreads();
  }
}

// ------------- K3: WMMA tensor-core reduction of partials ------------------
// A (16x32 f16) holds partials: row m -> quantity q=m%5, chunk=m/5 (m<15).
// B = ones => D[m][n] = sum_K A[m][K] accumulated in f32 C.
// Per iteration each quantity consumes 96 partials; 11 iters * 96 == 1056.
// Branchless gather: dead row (m==15) reads base 0 (in-bounds) scaled by 0,
// so EXEC stays all-ones through the WMMA loop and loads are 4x b128/lane.
__global__ void yolo_reduce_wmma_kernel(const float* __restrict__ part,
                                        float* __restrict__ out) {
  const int P = NBLK;                          // 1056 == 11*96
  const int lane = (int)threadIdx.x;           // 32 lanes, one wave
  const int m  = lane & 15;
  const int hi = (lane >> 4) & 1;
  const int q  = m % 5;
  const int chunk = m / 5;                     // 0..2 valid (m<15)
  const bool rowvalid = (m < 15);
  const float rv = rowvalid ? 1.0f : 0.0f;

  v16h ones;
  #pragma unroll
  for (int e = 0; e < 16; ++e) ones[e] = (_Float16)1.0f;

  v8f c = {};
  #pragma unroll
  for (int it = 0; it < 11; ++it) {
    const int base = rowvalid ? (it*96 + chunk*32 + hi*16) : 0;  // 16B-aligned
    const v4f_* src = (const v4f_*)(part + q*P + base);
    v4f_ f0 = src[0], f1 = src[1], f2 = src[2], f3 = src[3];
    v16h av;
    #pragma unroll
    for (int e = 0; e < 4; ++e) {
      av[e]      = (_Float16)(f0[e] * rv);
      av[4 + e]  = (_Float16)(f1[e] * rv);
      av[8 + e]  = (_Float16)(f2[e] * rv);
      av[12 + e] = (_Float16)(f3[e] * rv);
    }
    c = __builtin_amdgcn_wmma_f32_16x16x32_f16(
        /*neg_a=*/false, av, /*neg_b=*/false, ones,
        /*c_mod=*/(short)0, c, /*reuse_a=*/false, /*reuse_b=*/false);
  }

  // C layout: VGPR r, lanes 0-15 -> M=r,N=lane; lanes 16-31 -> M=r+8.
  float rowsum[16];
  #pragma unroll
  for (int r = 0; r < 8; ++r) {
    rowsum[r]     = __shfl(c[r], 0, 32);    // M=r,  N=0
    rowsum[r + 8] = __shfl(c[r], 16, 32);   // M=r+8,N=0
  }

  if (lane == 0) {
    float tq[5];
    #pragma unroll
    for (int qq = 0; qq < 5; ++qq)
      tq[qq] = rowsum[qq] + rowsum[qq + 5] + rowsum[qq + 10];
    const float conf_l = tq[0], conf_m = tq[1];
    const float loc_s  = tq[2], cls_s  = tq[3], n = tq[4];
    float loss = 0.0f;
    if (n > 0.0f) {
      loss = (loc_s / fmaxf(n, 1.0f)) * 0.05f            // BOX_RATIO
           + (cls_s / fmaxf(n * (float)NC, 1.0f)) * 1.0f; // CLS_RATIO
    }
    loss += (conf_l / fmaxf(conf_m, 1.0f)) * 20.0f;       // BALANCE[2]*OBJ_RATIO
    out[0] = loss;
  }
}

// ---------------------------------------------------------------------------
extern "C" void kernel_launch(void* const* d_in, const int* in_sizes, int n_in,
                              void* d_out, int out_size, void* d_ws, size_t ws_size,
                              hipStream_t stream) {
  (void)in_sizes; (void)n_in; (void)out_size; (void)ws_size;
  // d_in: [0]=ind (==2, folded into constants), [1]=input f32,
  //       [2]=targets_boxes f32, [3]=targets_cls i32
  const float* input  = (const float*)d_in[1];
  const float* tboxes = (const float*)d_in[2];
  const int*   tcls   = (const int*)d_in[3];
  float* out = (float*)d_out;

  int*   objmap = (int*)d_ws;
  float* part   = (float*)((char*)d_ws + sizeof(int)*(size_t)BSZ*NA*GHW);

  (void)hipMemsetAsync(objmap, 0, sizeof(int)*(size_t)BSZ*NA*GHW, stream);
  yolo_match_kernel<<<(BSZ*NT + 255)/256, 256, 0, stream>>>(tboxes, objmap);
  yolo_cell_kernel<<<NBLK, TILE, 0, stream>>>(input, tboxes, tcls, objmap, part);
  yolo_reduce_wmma_kernel<<<1, 32, 0, stream>>>(part, out);
}
